// Predictor_24034636988958
// MI455X (gfx1250) — matrix operand order repaired and checked by reading, not compiled, not run on hardware
//
#include <hip/hip_runtime.h>
#include <hip/hip_bf16.h>
#include <math.h>

#define BB 4
#define LL 256
#define QQ 128
#define HH 1024
#define NLAB 33
#define P_PAIRS 32896  // L*(L+1)/2

typedef __attribute__((ext_vector_type(16))) __bf16 v16bf;
typedef __attribute__((ext_vector_type(8)))  float  v8f;

union Frag { v16bf v; uint4 q[2]; };

// ---------------------------------------------------------------- conversions
__global__ void cvt_f32_bf16(const float* __restrict__ s, __bf16* __restrict__ d, int n) {
  int i = blockIdx.x * blockDim.x + threadIdx.x;
  if (i < n) d[i] = (__bf16)s[i];
}

// W (K x N, f32, row major) -> Wt (Npad x K, bf16, row major); rows >= N zero-filled.
__global__ void transpose_f32_bf16(const float* __restrict__ W, __bf16* __restrict__ Wt,
                                   int K, int N, int Npad) {
  __shared__ float tile[32][33];
  int k0 = blockIdx.x * 32, n0 = blockIdx.y * 32;
  for (int t = threadIdx.y; t < 32; t += 8) {
    int k = k0 + t, n = n0 + threadIdx.x;
    tile[t][threadIdx.x] = (k < K && n < N) ? W[(long)k * N + n] : 0.f;
  }
  __syncthreads();
  for (int t = threadIdx.y; t < 32; t += 8) {
    int np = n0 + t, k = k0 + threadIdx.x;
    if (np < Npad && k < K) Wt[(long)np * K + k] = (__bf16)tile[threadIdx.x][t];
  }
}

__global__ void pad_bias(const float* __restrict__ s, float* __restrict__ d, int n, int npad) {
  int i = blockIdx.x * blockDim.x + threadIdx.x;
  if (i < npad) d[i] = (i < n) ? s[i] : 0.f;
}

// ---------------------------------------------------------------- bf16 NT GEMM
// One wave computes a 32x32 macro-tile = 2x2 WMMA tiles: each A/B fragment is
// reused twice -> 2 b128 loads per v_wmma instead of 4 (doubles FLOP/byte).
// flags: 1=bias, 2=gelu, 4=store bf16 (else f32), 8=mask columns to -1e8
#define GF_BIAS 1
#define GF_GELU 2
#define GF_OUTBF 4
#define GF_MASK 8

#define WMMA_BF16(a, b, c) \
  __builtin_amdgcn_wmma_f32_16x16x32_bf16(false, (a).v, false, (b).v, (short)0, (c), false, false)

__global__ void __launch_bounds__(256)
gemm_bf16_wmma(const __bf16* __restrict__ A, int lda,
               const __bf16* __restrict__ Bt, int ldb,
               const float* __restrict__ bias,
               void* __restrict__ Cout, int ldc,
               int M, int N, int K, int nStore,
               float scale, int flags,
               const unsigned char* __restrict__ mask) {
  const int wave = threadIdx.x >> 5;
  const int lane = threadIdx.x & 31;
  const int tilesN = N >> 5;                  // 32-wide macro-tiles
  const int tiles  = (M >> 5) * tilesN;
  const int tile   = blockIdx.x * (blockDim.x >> 5) + wave;
  if (tile >= tiles) return;                  // wave-uniform: EXEC all-1 for WMMA
  const int tm = tile / tilesN, tn = tile % tilesN;
  const int g = lane >> 4, r = lane & 15;
  // A-frag: lane holds row r, K = {g*8..g*8+7, 16+g*8..16+g*8+7} (ISA 16-bit A layout)
  const __bf16* aPtr0 = A + (long)(tm * 32 + r) * lda + g * 8;
  const __bf16* aPtr1 = aPtr0 + (long)16 * lda;
  // B-frag: lane holds column r of B (= row r of Bt), K = {g*8..g*8+7, 16+g*8..16+g*8+7}
  const __bf16* bPtr0 = Bt + (long)(tn * 32 + r) * ldb + g * 8;
  const __bf16* bPtr1 = bPtr0 + (long)16 * ldb;
  v8f acc00 = {}, acc01 = {}, acc10 = {}, acc11 = {};
  for (int k0 = 0; k0 < K; k0 += 32) {
    Frag a0, a1, b0, b1;
    a0.q[0] = *(const uint4*)(aPtr0 + k0);
    a0.q[1] = *(const uint4*)(aPtr0 + k0 + 16);
    a1.q[0] = *(const uint4*)(aPtr1 + k0);
    a1.q[1] = *(const uint4*)(aPtr1 + k0 + 16);
    b0.q[0] = *(const uint4*)(bPtr0 + k0);
    b0.q[1] = *(const uint4*)(bPtr0 + k0 + 16);
    b1.q[0] = *(const uint4*)(bPtr1 + k0);
    b1.q[1] = *(const uint4*)(bPtr1 + k0 + 16);
    acc00 = WMMA_BF16(a0, b0, acc00);
    acc01 = WMMA_BF16(a0, b1, acc01);
    acc10 = WMMA_BF16(a1, b0, acc10);
    acc11 = WMMA_BF16(a1, b1, acc11);
  }
  float*  Cf = (float*)Cout;
  __bf16* Cb = (__bf16*)Cout;
  auto store = [&](const v8f& acc, int sm, int sn) {
    const int col = tn * 32 + sn * 16 + r;
    if (col >= nStore) return;                // after all WMMAs; lane-divergent OK
    const float bv = (flags & GF_BIAS) ? bias[col] : 0.f;
    const bool dead = (flags & GF_MASK) && (mask[col] == 0);
#pragma unroll
    for (int v = 0; v < 8; ++v) {
      int row = tm * 32 + sm * 16 + g * 8 + v;  // ISA C/D layout: VGPR v -> M=v (+8 hi lanes)
      float x = acc[v] * scale + bv;
      if (flags & GF_GELU) x = 0.5f * x * (1.f + erff(x * 0.70710678118654752f));
      if (dead) x = -1.0e8f;
      if (flags & GF_OUTBF) Cb[(long)row * ldc + col] = (__bf16)x;
      else                  Cf[(long)row * ldc + col] = x;
    }
  };
  store(acc00, 0, 0);
  store(acc01, 0, 1);
  store(acc10, 1, 0);
  store(acc11, 1, 1);
}

// ---------------------------------------------------------------- label softmax
__global__ void labels_kernel(const float* __restrict__ qlog,
                              const float* __restrict__ clsLog,  // [BQ x 48] padded
                              float* __restrict__ out) {
  int rIdx = blockIdx.x * blockDim.x + threadIdx.x;
  if (rIdx >= BB * QQ) return;
  float v[NLAB];
  float m = -3.4e38f;
  for (int t = 0; t < NLAB; ++t) {
    v[t] = qlog[rIdx * NLAB + t] + clsLog[rIdx * 48 + t];
    m = fmaxf(m, v[t]);
  }
  float s = 0.f;
  for (int t = 0; t < NLAB; ++t) { v[t] = __expf(v[t] - m); s += v[t]; }
  float inv = 1.f / s;
  for (int t = 0; t < NLAB; ++t) out[rIdx * NLAB + t] = v[t] * inv;
}

// ---------------------------------------------------------------- loc params: centers + fac@facT
__global__ void loc_params_kernel(const float* __restrict__ locOut,  // [BQ x 16]
                                  const float* __restrict__ qlocs,   // [BQ x 2]
                                  float* __restrict__ lp) {          // [BQ x 8]
  int rIdx = blockIdx.x * blockDim.x + threadIdx.x;
  if (rIdx >= BB * QQ) return;
  const float* o = locOut + rIdx * 16;
  float cx = qlocs[rIdx * 2 + 0] + o[0];
  float cy = qlocs[rIdx * 2 + 1] + o[1];
  float f0 = o[2], f1 = o[3], f2 = o[4], f3 = o[5];
  lp[rIdx * 8 + 0] = cx;
  lp[rIdx * 8 + 1] = cy;
  lp[rIdx * 8 + 2] = f0 * f0 + f1 * f1;   // F00
  lp[rIdx * 8 + 3] = f0 * f2 + f1 * f3;   // F01 = F10
  lp[rIdx * 8 + 4] = f2 * f2 + f3 * f3;   // F11
}

// ---------------------------------------------------------------- locs softmax over P pairs
__global__ void __launch_bounds__(256)
locs_kernel(const float* __restrict__ score,           // [B*2, Q, L] masked, /FACTOR
            const float* __restrict__ lp,              // [BQ x 8]
            const unsigned char* __restrict__ masks,   // [B, L]
            float* __restrict__ outLocs) {             // [BQ, P_PAIRS]
  __shared__ float s0[LL], s1[LL], mk[LL];
  __shared__ int off[LL + 1];
  __shared__ float red[256];
  const int tid = threadIdx.x;
  const int bq = blockIdx.x;
  const int b = bq / QQ, q = bq % QQ;
  const float* s0g = score + ((long)(b * 2 + 0) * QQ + q) * LL;
  const float* s1g = score + ((long)(b * 2 + 1) * QQ + q) * LL;
  for (int v = tid; v < LL; v += 256) {
    s0[v] = s0g[v];
    s1[v] = s1g[v];
    mk[v] = masks[b * LL + v] ? 1.f : 0.f;
  }
  for (int i = tid; i <= LL; i += 256) off[i] = i * LL - (i * (i - 1)) / 2;
  __syncthreads();
  const float cx = lp[bq * 8 + 0], cy = lp[bq * 8 + 1];
  const float fA = lp[bq * 8 + 2], fB = lp[bq * 8 + 3], fC = lp[bq * 8 + 4];

  // pass 1: m = max_p score_p
  float lmax = -3.4e38f;
  for (int p = tid; p < P_PAIRS; p += 256) {
    int lo = 0, hi = LL - 1;
    while (lo < hi) { int mid = (lo + hi + 1) >> 1; if (off[mid] <= p) lo = mid; else hi = mid - 1; }
    int j = lo + (p - off[lo]);
    lmax = fmaxf(lmax, s0[lo] + s1[j]);
  }
  red[tid] = lmax; __syncthreads();
  for (int s = 128; s > 0; s >>= 1) { if (tid < s) red[tid] = fmaxf(red[tid], red[tid + s]); __syncthreads(); }
  const float m = red[0];
  __syncthreads();

  // pass 2: e = mask_j * exp(score_p - m - quad), store + sum
  float* orow = outLocs + (long)bq * P_PAIRS;
  float lsum = 0.f;
  for (int p = tid; p < P_PAIRS; p += 256) {
    int lo = 0, hi = LL - 1;
    while (lo < hi) { int mid = (lo + hi + 1) >> 1; if (off[mid] <= p) lo = mid; else hi = mid - 1; }
    int i = lo, j = i + (p - off[i]);
    float sp = s0[i] + s1[j];
    float dx = (float)i - cx, dy = (float)j - cy;
    float quad = fA * dx * dx + 2.f * fB * dx * dy + fC * dy * dy;
    float e = mk[j] * __expf(sp - m - quad);   // exponent <= 0: no overflow
    orow[p] = e;
    lsum += e;
  }
  red[tid] = lsum; __syncthreads();
  for (int s = 128; s > 0; s >>= 1) { if (tid < s) red[tid] += red[tid + s]; __syncthreads(); }
  const float inv = 1.f / (red[0] + 1e-12f);
  // pass 3: normalize (re-read stays in 192MB L2)
  for (int p = tid; p < P_PAIRS; p += 256) orow[p] *= inv;
}

// ---------------------------------------------------------------- host
extern "C" void kernel_launch(void* const* d_in, const int* in_sizes, int n_in,
                              void* d_out, int out_size, void* d_ws, size_t ws_size,
                              hipStream_t stream) {
  (void)in_sizes; (void)n_in; (void)out_size; (void)ws_size;
  const float* hiddens = (const float*)d_in[0];
  const unsigned char* masks = (const unsigned char*)d_in[1];  // bool array
  const float* queries = (const float*)d_in[2];
  const float* qlocs   = (const float*)d_in[3];
  const float* qlogits = (const float*)d_in[4];
  const float* cls_w1  = (const float*)d_in[5];
  const float* cls_b1  = (const float*)d_in[6];
  const float* cls_w2  = (const float*)d_in[7];
  const float* cls_b2  = (const float*)d_in[8];
  const float* tq_w    = (const float*)d_in[9];
  const float* tq_b    = (const float*)d_in[10];
  const float* tc_w    = (const float*)d_in[11];
  const float* tc_b    = (const float*)d_in[12];
  const float* loc_w1  = (const float*)d_in[13];
  const float* loc_b1  = (const float*)d_in[14];
  const float* loc_w2  = (const float*)d_in[15];
  const float* loc_b2  = (const float*)d_in[16];

  char* ws = (char*)d_ws;
  size_t off = 0;
  auto alloc = [&](size_t bytes) -> void* {
    void* p = (void*)(ws + off);
    off += (bytes + 255) & ~(size_t)255;
    return p;
  };
  __bf16* Qb    = (__bf16*)alloc((size_t)512 * 1024 * 2);
  __bf16* Hb    = (__bf16*)alloc((size_t)1024 * 1024 * 2);
  __bf16* tqwT  = (__bf16*)alloc((size_t)2048 * 1024 * 2);
  __bf16* tcwT  = (__bf16*)alloc((size_t)2048 * 1024 * 2);
  __bf16* cw1T  = (__bf16*)alloc((size_t)512 * 1024 * 2);
  __bf16* lw1T  = (__bf16*)alloc((size_t)512 * 1024 * 2);
  __bf16* cw2T  = (__bf16*)alloc((size_t)64 * 512 * 2);   // N=33 padded to 64
  __bf16* lw2T  = (__bf16*)alloc((size_t)32 * 512 * 2);   // N=6  padded to 32
  __bf16* QHb   = (__bf16*)alloc((size_t)512 * 2048 * 2);
  __bf16* CHb   = (__bf16*)alloc((size_t)1024 * 2048 * 2);
  __bf16* clsHb = (__bf16*)alloc((size_t)512 * 512 * 2);
  __bf16* locHb = (__bf16*)alloc((size_t)512 * 512 * 2);
  float*  scoreW = (float*)alloc((size_t)8 * QQ * LL * 4);
  float*  clsLog = (float*)alloc((size_t)512 * 48 * 4);
  float*  locOut = (float*)alloc((size_t)512 * 16 * 4);
  float*  lpW    = (float*)alloc((size_t)512 * 8 * 4);
  float*  cb2p   = (float*)alloc(64 * 4);
  float*  lb2p   = (float*)alloc(32 * 4);

  // activations -> bf16
  cvt_f32_bf16<<<(512 * 1024 + 255) / 256, 256, 0, stream>>>(queries, Qb, 512 * 1024);
  cvt_f32_bf16<<<(1024 * 1024 + 255) / 256, 256, 0, stream>>>(hiddens, Hb, 1024 * 1024);
  // weights: transpose + cast (zero-pad narrow N to multiples of 32)
  dim3 tb(32, 8);
  transpose_f32_bf16<<<dim3(32, 64), tb, 0, stream>>>(tq_w, tqwT, 1024, 2048, 2048);
  transpose_f32_bf16<<<dim3(32, 64), tb, 0, stream>>>(tc_w, tcwT, 1024, 2048, 2048);
  transpose_f32_bf16<<<dim3(32, 16), tb, 0, stream>>>(cls_w1, cw1T, 1024, 512, 512);
  transpose_f32_bf16<<<dim3(32, 16), tb, 0, stream>>>(loc_w1, lw1T, 1024, 512, 512);
  transpose_f32_bf16<<<dim3(16, 2),  tb, 0, stream>>>(cls_w2, cw2T, 512, 33, 64);
  transpose_f32_bf16<<<dim3(16, 1),  tb, 0, stream>>>(loc_w2, lw2T, 512, 6, 32);
  pad_bias<<<1, 64, 0, stream>>>(cls_b2, cb2p, 33, 64);
  pad_bias<<<1, 64, 0, stream>>>(loc_b2, lb2p, 6, 32);

  auto gemm = [&](const __bf16* A, int lda, const __bf16* Bt, int ldb, const float* bias,
                  void* C, int ldc, int M, int N, int K, int nStore, float scale, int flags,
                  const unsigned char* mask) {
    int tiles = (M / 32) * (N / 32);
    int blocks = (tiles + 7) / 8;
    gemm_bf16_wmma<<<blocks, 256, 0, stream>>>(A, lda, Bt, ldb, bias, C, ldc,
                                               M, N, K, nStore, scale, flags, mask);
  };

  // projections (bf16 out for score GEMM)
  gemm(Qb, 1024, tqwT, 1024, tq_b, QHb, 2048, 512, 2048, 1024, 2048, 1.f, GF_BIAS | GF_OUTBF, nullptr);
  gemm(Hb, 1024, tcwT, 1024, tc_b, CHb, 2048, 1024, 2048, 1024, 2048, 1.f, GF_BIAS | GF_OUTBF, nullptr);
  // cls / loc MLPs
  gemm(Qb, 1024, cw1T, 1024, cls_b1, clsHb, 512, 512, 512, 1024, 512, 1.f,
       GF_BIAS | GF_GELU | GF_OUTBF, nullptr);
  gemm(Qb, 1024, lw1T, 1024, loc_b1, locHb, 512, 512, 512, 1024, 512, 1.f,
       GF_BIAS | GF_GELU | GF_OUTBF, nullptr);
  gemm(clsHb, 512, cw2T, 512, cb2p, clsLog, 48, 512, 64, 512, 33, 1.f, GF_BIAS, nullptr);
  gemm(locHb, 512, lw2T, 512, lb2p, locOut, 16, 512, 32, 512, 16, 1.f, GF_BIAS, nullptr);
  // score[b,c] = (QH_c · CH_cᵀ)/32, masked to -1e8 where key invalid
  for (int b = 0; b < BB; ++b)
    for (int c = 0; c < 2; ++c)
      gemm(QHb + (size_t)b * QQ * 2048 + c * 1024, 2048,
           CHb + (size_t)b * LL * 2048 + c * 1024, 2048, nullptr,
           scoreW + (size_t)(b * 2 + c) * QQ * LL, LL,
           QQ, LL, 1024, LL, 1.f / 32.f, GF_MASK, masks + b * LL);

  float* outLabels = (float*)d_out;
  float* outLocs   = (float*)d_out + (size_t)BB * QQ * NLAB;
  labels_kernel<<<2, 256, 0, stream>>>(qlogits, clsLog, outLabels);
  loc_params_kernel<<<2, 256, 0, stream>>>(locOut, qlocs, lpW);
  locs_kernel<<<BB * QQ, 256, 0, stream>>>(scoreW, lpW, masks, outLocs);
}
